// GNN_8572754723292
// MI455X (gfx1250) — matrix operand order repaired and checked by reading, not compiled, not run on hardware
//
#include <hip/hip_runtime.h>
#include <math.h>

typedef __attribute__((ext_vector_type(16))) _Float16 v16h;
typedef __attribute__((ext_vector_type(8)))  float    v8f;

#define FDIM 16

// ================= degree / norm precompute =================
__global__ void k_deg_init(float* deg, int n) {
    int i = blockIdx.x * blockDim.x + threadIdx.x;
    if (i < n) deg[i] = 1.0f;                       // self loop
}

// direct path: count only
__global__ void k_deg_count(const long long* __restrict__ ei, float* deg, int E) {
    int e = blockIdx.x * blockDim.x + threadIdx.x;
    if (e < E) unsafeAtomicAdd(&deg[(int)ei[(size_t)E + e]], 1.0f);
}

// prep path: count + int64 -> int32 conversion (indices reused 3x per run)
__global__ void k_prep(const long long* __restrict__ ei, int* __restrict__ s32,
                       int* __restrict__ d32, float* deg, int E) {
    int e = blockIdx.x * blockDim.x + threadIdx.x;
    if (e < E) {
        int s = (int)ei[e];
        int d = (int)ei[(size_t)E + e];
        s32[e] = s;
        d32[e] = d;
        unsafeAtomicAdd(&deg[d], 1.0f);
    }
}

__global__ void k_deg_finish(float* deg, int n) {
    int i = blockIdx.x * blockDim.x + threadIdx.x;
    if (i < n) deg[i] = rsqrtf(deg[i]);             // deg >= 1 always
}

__global__ void k_norm(const int* __restrict__ s32, const int* __restrict__ d32,
                       const float* __restrict__ dinv, float* __restrict__ nrm, int E) {
    int e = blockIdx.x * blockDim.x + threadIdx.x;
    if (e < E) nrm[e] = dinv[s32[e]] * dinv[d32[e]];
}

// ================= WMMA GEMM: H = X @ W, fused Agg = H*dinv^2 =================
// One wave32 = one 16-node tile. K padded 16 -> 32 with zeros.
__global__ void k_gcn_gemm(const float* __restrict__ X, const float* __restrict__ W,
                           const float* __restrict__ dinv,
                           float* __restrict__ H, float* __restrict__ Agg, int N) {
    int wave = threadIdx.x >> 5;
    int lane = threadIdx.x & 31;
    int tile = blockIdx.x * 8 + wave;
    if (tile * 16 >= N) return;                     // wave-uniform, no barriers here

    int m    = lane & 15;
    int half = lane >> 4;
    int koff = half * 8;
    int node = tile * 16 + m;

    // A: 16x32 f16 (VGPRs 0-3 real K, 4-7 zero pad)
    v16h a = (v16h)(_Float16)0.0f;
    if (node < N) {
        const float4* xr = (const float4*)(X + (size_t)node * FDIM + koff);
        float4 p0 = xr[0], p1 = xr[1];
        a[0] = (_Float16)p0.x; a[1] = (_Float16)p0.y; a[2] = (_Float16)p0.z; a[3] = (_Float16)p0.w;
        a[4] = (_Float16)p1.x; a[5] = (_Float16)p1.y; a[6] = (_Float16)p1.z; a[7] = (_Float16)p1.w;
    }

    // B: 32x16 f16. Lane half h carries K=8h..8h+7 of col n in elements 0-7; elements 8-15 (K pad) zero.
    int n = lane & 15;
    v16h b = (v16h)(_Float16)0.0f;
    {
        #pragma unroll
        for (int k = 0; k < 8; ++k) b[k] = (_Float16)W[(koff + k) * 16 + n];
    }

    v8f c = {};
    c = __builtin_amdgcn_wmma_f32_16x16x32_f16(false, a, false, b, (short)0, c, false, false);

    int mbase = half * 8;
    if (tile * 16 + 16 <= N) {                      // uniform full-tile fast path
        #pragma unroll
        for (int r = 0; r < 8; ++r) {
            int nd = tile * 16 + mbase + r;
            float dv = dinv[nd];
            size_t off = (size_t)nd * FDIM + n;
            H[off]   = c[r];
            Agg[off] = c[r] * dv * dv;              // self-loop term
        }
    } else {
        #pragma unroll
        for (int r = 0; r < 8; ++r) {
            int nd = tile * 16 + mbase + r;
            if (nd < N) {
                float dv = dinv[nd];
                size_t off = (size_t)nd * FDIM + n;
                H[off]   = c[r];
                Agg[off] = c[r] * dv * dv;
            }
        }
    }
}

// ================= edge scatter (4 lanes/edge, b128 gather, hw f32 atomics) =================
__global__ void k_scatter_pre(const int* __restrict__ s32, const int* __restrict__ d32,
                              const float* __restrict__ nrm, const float* __restrict__ h,
                              float* __restrict__ agg, int E) {
    int t = blockIdx.x * blockDim.x + threadIdx.x;
    int e = t >> 2, q = t & 3;
    if (e >= E) return;
    int s = s32[e], d = d32[e];
    float nm = nrm[e];
    float4 hv = ((const float4*)(h + (size_t)s * FDIM))[q];
    float* ap = agg + (size_t)d * FDIM + q * 4;
    unsafeAtomicAdd(ap + 0, hv.x * nm);
    unsafeAtomicAdd(ap + 1, hv.y * nm);
    unsafeAtomicAdd(ap + 2, hv.z * nm);
    unsafeAtomicAdd(ap + 3, hv.w * nm);
}

__global__ void k_scatter_dir(const long long* __restrict__ ei, const float* __restrict__ dinv,
                              const float* __restrict__ h, float* __restrict__ agg, int E) {
    int t = blockIdx.x * blockDim.x + threadIdx.x;
    int e = t >> 2, q = t & 3;
    if (e >= E) return;
    int s = (int)ei[e];
    int d = (int)ei[(size_t)E + e];
    float nm = dinv[s] * dinv[d];
    float4 hv = ((const float4*)(h + (size_t)s * FDIM))[q];
    float* ap = agg + (size_t)d * FDIM + q * 4;
    unsafeAtomicAdd(ap + 0, hv.x * nm);
    unsafeAtomicAdd(ap + 1, hv.y * nm);
    unsafeAtomicAdd(ap + 2, hv.z * nm);
    unsafeAtomicAdd(ap + 3, hv.w * nm);
}

// ================= bias + tanh =================
__global__ void k_bias_tanh(const float* __restrict__ agg, const float* __restrict__ b,
                            float* __restrict__ act, int total) {
    int i = blockIdx.x * blockDim.x + threadIdx.x;
    if (i < total) act[i] = tanhf(agg[i] + b[i & 15]);
}

// ================= MLP head, fully WMMA: 16 -> 32 -> 64 -> 8 =================
// One wave32 = one 16-node tile; relu'd f16 activations restaged via per-wave LDS
// (row-major, so A-fragments are contiguous 16B reads -- no transpose needed).
__global__ void k_mlp_wmma(const float* __restrict__ act,
                           const float* __restrict__ wf,  const float* __restrict__ bf,
                           const float* __restrict__ wf2, const float* __restrict__ bf2,
                           const float* __restrict__ wf3, const float* __restrict__ bf3,
                           float* __restrict__ out, int N) {
    __shared__ _Float16 sH1[8][16 * 32];
    __shared__ _Float16 sH2[8][16 * 64];

    int wave = threadIdx.x >> 5;
    int lane = threadIdx.x & 31;
    int tile = blockIdx.x * 8 + wave;
    bool active = (tile * 16) < N;                  // NO early return: barriers below

    int m     = lane & 15;
    int half  = lane >> 4;
    int koff  = half * 8;
    int n     = lane & 15;
    int mbase = half * 8;
    int node  = tile * 16 + m;

    // ---- A0: input tile, K=16 padded to 32 ----
    v16h a0 = (v16h)(_Float16)0.0f;
    if (active && node < N) {
        const float4* xr = (const float4*)(act + (size_t)node * FDIM + koff);
        float4 p0 = xr[0], p1 = xr[1];
        a0[0] = (_Float16)p0.x; a0[1] = (_Float16)p0.y; a0[2] = (_Float16)p0.z; a0[3] = (_Float16)p0.w;
        a0[4] = (_Float16)p1.x; a0[5] = (_Float16)p1.y; a0[6] = (_Float16)p1.z; a0[7] = (_Float16)p1.w;
    }

    // ---- layer 1: 16 -> 32 (2 N-tiles) ----
    #pragma unroll
    for (int nt = 0; nt < 2; ++nt) {
        int col = nt * 16 + n;
        v16h b = (v16h)(_Float16)0.0f;
        {
            #pragma unroll
            for (int k = 0; k < 8; ++k) b[k] = (_Float16)wf[(koff + k) * 32 + col];
        }
        v8f c = {};
        c = __builtin_amdgcn_wmma_f32_16x16x32_f16(false, a0, false, b, (short)0, c, false, false);
        float bv = bf[col];
        #pragma unroll
        for (int r = 0; r < 8; ++r)
            sH1[wave][(mbase + r) * 32 + col] = (_Float16)fmaxf(c[r] + bv, 0.0f);
    }
    __syncthreads();

    // ---- A1: 16x32 from LDS (K=32, exact WMMA shape) ----
    v16h a1;
    #pragma unroll
    for (int j = 0; j < 8; ++j) {
        a1[j]     = sH1[wave][m * 32 + koff + j];
        a1[8 + j] = sH1[wave][m * 32 + 16 + koff + j];
    }

    // ---- layer 2: 32 -> 64 (4 N-tiles) ----
    #pragma unroll
    for (int nt = 0; nt < 4; ++nt) {
        int col = nt * 16 + n;
        v16h b;
        #pragma unroll
        for (int j = 0; j < 16; ++j) b[j] = (_Float16)wf2[((j & 8) * 2 + koff + (j & 7)) * 64 + col];
        v8f c = {};
        c = __builtin_amdgcn_wmma_f32_16x16x32_f16(false, a1, false, b, (short)0, c, false, false);
        float bv = bf2[col];
        #pragma unroll
        for (int r = 0; r < 8; ++r)
            sH2[wave][(mbase + r) * 64 + col] = (_Float16)fmaxf(c[r] + bv, 0.0f);
    }
    __syncthreads();

    // ---- A2/A3: 16x64 from LDS as two K=32 chunks ----
    v16h a2, a3;
    #pragma unroll
    for (int j = 0; j < 8; ++j) {
        a2[j]     = sH2[wave][m * 64 + koff + j];
        a2[8 + j] = sH2[wave][m * 64 + 16 + koff + j];
        a3[j]     = sH2[wave][m * 64 + 32 + koff + j];
        a3[8 + j] = sH2[wave][m * 64 + 48 + koff + j];
    }

    // ---- layer 3: 64 -> 8 (N padded 8 -> 16), 2 accumulated WMMAs ----
    v16h b0 = (v16h)(_Float16)0.0f, b1 = (v16h)(_Float16)0.0f;
    if (n < 8) {
        #pragma unroll
        for (int j = 0; j < 16; ++j) {
            b0[j] = (_Float16)wf3[((j & 8) * 2 + koff + (j & 7)) * 8 + n];
            b1[j] = (_Float16)wf3[(32 + (j & 8) * 2 + koff + (j & 7)) * 8 + n];
        }
    }
    v8f c = {};
    c = __builtin_amdgcn_wmma_f32_16x16x32_f16(false, a2, false, b0, (short)0, c, false, false);
    c = __builtin_amdgcn_wmma_f32_16x16x32_f16(false, a3, false, b1, (short)0, c, false, false);

    if (active && n < 8) {
        float bv = bf3[n];
        #pragma unroll
        for (int r = 0; r < 8; ++r) {
            int nd = tile * 16 + mbase + r;
            if (nd < N) out[(size_t)nd * 8 + n] = c[r] + bv;
        }
    }
}

// ================= launch =================
extern "C" void kernel_launch(void* const* d_in, const int* in_sizes, int n_in,
                              void* d_out, int out_size, void* d_ws, size_t ws_size,
                              hipStream_t stream) {
    const float*     x  = (const float*)d_in[0];
    const long long* ei = (const long long*)d_in[1];
    const float* W[3] = {(const float*)d_in[2], (const float*)d_in[4], (const float*)d_in[6]};
    const float* B[3] = {(const float*)d_in[3], (const float*)d_in[5], (const float*)d_in[7]};
    const float* wf  = (const float*)d_in[8];
    const float* bf  = (const float*)d_in[9];
    const float* wf2 = (const float*)d_in[10];
    const float* bf2 = (const float*)d_in[11];
    const float* wf3 = (const float*)d_in[12];
    const float* bf3 = (const float*)d_in[13];

    int N = in_sizes[0] / FDIM;
    int E = in_sizes[1] / 2;
    size_t N4 = ((size_t)N + 3) & ~(size_t)3;       // keep 16B alignment downstream

    float* ws   = (float*)d_ws;
    float* dinv = ws;                               // N4 floats
    float* H    = dinv + N4;                        // 16N
    float* Agg  = H    + (size_t)N * FDIM;          // 16N
    float* Act  = Agg  + (size_t)N * FDIM;          // 16N
    int*   s32  = (int*)(Act + (size_t)N * FDIM);   // E ints   (optional)
    int*   d32  = s32 + E;                          // E ints   (optional)
    float* nrm  = (float*)(d32 + E);                // E floats (optional)

    size_t needExt = (N4 + 48ull * N) * 4 + (size_t)E * 12;
    bool prep = (ws_size >= needExt);

    const int TB = 256;
    int nb_n   = (N + TB - 1) / TB;
    int nb_e   = (E + TB - 1) / TB;
    int nb_nf  = (N * FDIM + TB - 1) / TB;
    int nb_e4  = (int)(((long long)E * 4 + TB - 1) / TB);
    int nb_gem = ((N + 15) / 16 + 7) / 8;           // 8 wave-tiles per block

    // normalization coefficients (+ optional edge precompute)
    k_deg_init<<<nb_n, TB, 0, stream>>>(dinv, N);
    if (prep) k_prep<<<nb_e, TB, 0, stream>>>(ei, s32, d32, dinv, E);
    else      k_deg_count<<<nb_e, TB, 0, stream>>>(ei, dinv, E);
    k_deg_finish<<<nb_n, TB, 0, stream>>>(dinv, N);
    if (prep) k_norm<<<nb_e, TB, 0, stream>>>(s32, d32, dinv, nrm, E);

    // three GCN layers
    for (int l = 0; l < 3; ++l) {
        const float* in = (l == 0) ? x : Act;
        k_gcn_gemm<<<nb_gem, TB, 0, stream>>>(in, W[l], dinv, H, Agg, N);
        if (prep) k_scatter_pre<<<nb_e4, TB, 0, stream>>>(s32, d32, nrm, H, Agg, E);
        else      k_scatter_dir<<<nb_e4, TB, 0, stream>>>(ei, dinv, H, Agg, E);
        k_bias_tanh<<<nb_nf, TB, 0, stream>>>(Agg, B[l], Act, N * FDIM);
    }

    // WMMA MLP head
    k_mlp_wmma<<<nb_gem, TB, 0, stream>>>(Act, wf, bf, wf2, bf2, wf3, bf3, (float*)d_out, N);
}
